// WindowAttention_10264971837500
// MI455X (gfx1250) — matrix operand order repaired and checked by reading, not compiled, not run on hardware
//
#include <hip/hip_runtime.h>

typedef __attribute__((ext_vector_type(16))) _Float16 v16h;
typedef __attribute__((ext_vector_type(8)))  _Float16 v8h;
typedef __attribute__((ext_vector_type(8)))  float    v8f;

// ---------------- problem constants ----------------
#define IMG    512
#define CCH    128
#define WINSZ  8
#define NTOK   64        // tokens per window
#define NHEAD  4
#define HDIM   32
#define SHIFTV 4
#define QKV_N  384
// repacked weight sizes (halfs)
#define QKV_HALF (24*4*32*16)   // 49152: 24 n-tiles x 4 k-steps x 32 lanes x 16 halfs
#define PRJ_HALF (8*4*32*16)    // 16384

// ---------------- WMMA helper ----------------
__device__ __forceinline__ v8f wmma_f16(v16h a, v16h b, v8f c) {
  // D = A(16x32 f16) * B(32x16 f16) + C(16x16 f32)
  return __builtin_amdgcn_wmma_f32_16x16x32_f16(false, a, false, b, (short)0, c,
                                                false, false);
}

// A-fragment (16x32 f16) from a row-major f16 buffer in LDS.
// Layout: lanes 0-15 row M=lane hold K = kk+[0..15];
// lanes 16-31 row M=lane-16 hold K = kk+[16..31].
__device__ __forceinline__ v16h load_a_frag(const _Float16* rowptr, int kk, int lane) {
  const int koff = (lane < 16) ? 0 : 16;
  v8h lo = *(const v8h*)(rowptr + kk + koff);
  v8h hi = *(const v8h*)(rowptr + kk + 8 + koff);
  v16h a;
#pragma unroll
  for (int j = 0; j < 8; ++j) { a[j] = lo[j]; a[8 + j] = hi[j]; }
  return a;
}

// ---------------- weight repack: f32 row-major -> f16 B-fragment stream ------
// B-fragment layout (32x16 f16): lane = n0+(lane&15) column;
// lanes 0-15 elements e hold K=kk+e (e=0..15), lanes 16-31 hold K=kk+16+e.
// Stored so that a wave's fragment is 1KB contiguous; each lane reads 32B.
__global__ void repack_weights(const float* __restrict__ qkv_w,
                               const float* __restrict__ proj_w,
                               _Float16* __restrict__ qkvB,
                               _Float16* __restrict__ projB) {
  int idx = blockIdx.x * blockDim.x + threadIdx.x;
  if (idx < QKV_HALF) {
    int e    = idx & 15;
    int lane = (idx >> 4) & 31;
    int ks   = (idx >> 9) & 3;
    int nt   = idx >> 11;                  // 0..23
    int k    = ks * 32 + ((lane < 16) ? 0 : 16) + e;
    int n    = nt * 16 + (lane & 15);
    qkvB[idx] = (_Float16)qkv_w[k * QKV_N + n];
  } else if (idx < QKV_HALF + PRJ_HALF) {
    int j    = idx - QKV_HALF;
    int e    = j & 15;
    int lane = (j >> 4) & 31;
    int ks   = (j >> 9) & 3;
    int nt   = j >> 11;                    // 0..7
    int k    = ks * 32 + ((lane < 16) ? 0 : 16) + e;
    int n    = nt * 16 + (lane & 15);
    projB[j] = (_Float16)proj_w[k * CCH + n];
  }
}

// ---------------- fused shifted-window attention -----------------------------
// One workgroup (8 wave32) = one 8x8 window. All intermediates in LDS (f16),
// WMMA f16 with f32 accumulation throughout.
//
// LDS overlay (exactly 64KB = 32768 halfs):
//   [    0, 8192)  xs  [64][128]   ph1-2   | ps [4][64][64] ph3b-4 | outs f32 ph5
//   [ 8192,16384)  qs  [4][64][32] ph2-3a  | (ps upper half / outs upper half)
//   [16384,24576)  ksd [4][64][32] ph2-3a  | aos [64][128] ph4-5
//   [24576,32768)  vts [4][32][64] ph2-4
__launch_bounds__(256, 1)
__global__ void swin_attn_fused(const float* __restrict__ x,
                                const float* __restrict__ qkv_b,
                                const float* __restrict__ proj_b,
                                const float* __restrict__ bias_tab,
                                const _Float16* __restrict__ qkvB,
                                const _Float16* __restrict__ projB,
                                float* __restrict__ out) {
  __shared__ __align__(16) _Float16 smem[32768];
  _Float16* xs  = smem;                 // [64][128]
  _Float16* qs  = smem + 8192;          // [4][64][32]
  _Float16* ksd = smem + 16384;         // [4][64][32]
  _Float16* vts = smem + 24576;         // [4][32][64]
  _Float16* ps  = smem;                 // [4][64][64] (overlays xs+qs)
  _Float16* aos = smem + 16384;         // [64][128]   (overlays ksd)
  float*    outs = reinterpret_cast<float*>(smem);   // [64][128] f32 (overlays ps)

  const int tid  = threadIdx.x;
  const int lane = tid & 31;
  const int wave = tid >> 5;
  const int win  = blockIdx.x;
  const int wx   = win & 63;
  const int wy   = (win >> 6) & 63;
  const int bb   = win >> 12;
  const int l15  = lane & 15;
  const int lhi8 = (lane >> 4) << 3;         // 0 or 8: M offset of C rows
  const int koff2 = (lane < 16) ? 0 : 16;    // B-frag K offset

  // ---- phase 1: gather shifted window, f32 -> f16 (roll(-S): src=(g+S)&511)
  for (int i = tid; i < NTOK * (CCH / 4); i += 256) {
    int t  = i >> 5;                  // token
    int c4 = i & 31;                  // float4 within channel row
    int gy = (wy * WINSZ + (t >> 3) + SHIFTV) & (IMG - 1);
    int gx = (wx * WINSZ + (t & 7) + SHIFTV) & (IMG - 1);
    const float4 v =
        *(const float4*)(x + ((((size_t)bb * IMG + gy) * IMG + gx) * CCH) + c4 * 4);
    _Float16* d = xs + t * CCH + c4 * 4;
    d[0] = (_Float16)v.x; d[1] = (_Float16)v.y;
    d[2] = (_Float16)v.z; d[3] = (_Float16)v.w;
  }
  __syncthreads();

  // ---- phase 2: qkv = xs @ Wqkv + b   (96 tiles, K=128 in 4 steps)
  {
    const int mt = wave & 3;          // fixed M-tile per wave -> A-frags hoisted
    const int m0 = mt * 16;
    const _Float16* arow = xs + (m0 + l15) * CCH;
    const v16h a0 = load_a_frag(arow, 0,  lane);
    const v16h a1 = load_a_frag(arow, 32, lane);
    const v16h a2 = load_a_frag(arow, 64, lane);
    const v16h a3 = load_a_frag(arow, 96, lane);
    for (int i = 0; i < 12; ++i) {
      const int nt = (wave >> 2) + 2 * i;   // 0..23 across waves
      const _Float16* bp = qkvB + (((size_t)nt * 4) * 32 + lane) * 16;
      v8f acc = {};
      acc = wmma_f16(a0, *(const v16h*)(bp + 0 * 512), acc);
      acc = wmma_f16(a1, *(const v16h*)(bp + 1 * 512), acc);
      acc = wmma_f16(a2, *(const v16h*)(bp + 2 * 512), acc);
      acc = wmma_f16(a3, *(const v16h*)(bp + 3 * 512), acc);
      if (i + 1 < 12)  // prefetch next fragment stream (global_prefetch_b8)
        __builtin_prefetch(bp + 4 * 512, 0, 3);

      // branchless scatter into qs / ksd / vts (contiguous LDS regions)
      const int   n     = nt * 16 + l15;     // 0..383
      const float biasn = qkv_b[n];
      const int   three = n >> 7;            // 0=q 1=k 2=v
      const int   hh    = (n >> 5) & 3;
      const int   dd    = n & 31;
      const int   stride = (three == 2) ? 1 : HDIM;
      int base = 8192 + (three << 13)
               + ((three == 2) ? ((hh * HDIM + dd) * NTOK)
                               : (hh * NTOK * HDIM + dd))
               + (m0 + lhi8) * stride;
#pragma unroll
      for (int r = 0; r < 8; ++r)
        smem[base + r * stride] = (_Float16)(acc[r] + biasn);
    }
  }
  __syncthreads();

  // ---- phase 3a: scores = q @ k^T * scale + relpos_bias (all in registers)
  const int   h     = wave >> 1;
  const int   mhalf = wave & 1;          // which 32 query rows
  const float sc    = 0.17677669529663687f;  // 32^-0.5
  float sreg[2][4][8];
#pragma unroll
  for (int mt2 = 0; mt2 < 2; ++mt2) {
    const int m0 = mhalf * 32 + mt2 * 16;
    v16h a = load_a_frag(qs + (h * NTOK + m0 + l15) * HDIM, 0, lane);
#pragma unroll
    for (int nt = 0; nt < 4; ++nt) {
      const int m = nt * 16 + l15;       // key token (C column)
      v16h bf = *(const v16h*)(ksd + (h * NTOK + m) * HDIM + koff2);
      v8f acc = {};
      acc = wmma_f16(a, bf, acc);
#pragma unroll
      for (int r = 0; r < 8; ++r) {
        const int n    = m0 + r + lhi8;  // query token (C row)
        const int ridx = ((n >> 3) - (m >> 3) + 7) * 15 + ((n & 7) - (m & 7) + 7);
        sreg[mt2][nt][r] = acc[r] * sc + bias_tab[ridx * NHEAD + h];
      }
    }
  }
  __syncthreads();   // everyone done reading qs/ksd before ps/aos overlays

  // ---- phase 3b: softmax across 64 columns (4 reg tiles x 16 lanes) -> ps f16
#pragma unroll
  for (int mt2 = 0; mt2 < 2; ++mt2) {
    const int m0 = mhalf * 32 + mt2 * 16;
#pragma unroll
    for (int r = 0; r < 8; ++r) {
      float mx = fmaxf(fmaxf(sreg[mt2][0][r], sreg[mt2][1][r]),
                       fmaxf(sreg[mt2][2][r], sreg[mt2][3][r]));
#pragma unroll
      for (int msk = 1; msk < 16; msk <<= 1) mx = fmaxf(mx, __shfl_xor(mx, msk, 32));
      float p[4], sum = 0.f;
#pragma unroll
      for (int nt = 0; nt < 4; ++nt) { p[nt] = __expf(sreg[mt2][nt][r] - mx); sum += p[nt]; }
#pragma unroll
      for (int msk = 1; msk < 16; msk <<= 1) sum += __shfl_xor(sum, msk, 32);
      const float inv = 1.0f / sum;
      const int   n   = m0 + r + lhi8;
#pragma unroll
      for (int nt = 0; nt < 4; ++nt)
        ps[(h * NTOK + n) * NTOK + nt * 16 + l15] = (_Float16)(p[nt] * inv);
    }
  }
  __syncthreads();

  // ---- phase 4: out_h = P @ V  (K=64 in 2 steps), store f16 to aos
  {
    const int m0 = mhalf * 32;
#pragma unroll
    for (int mt2 = 0; mt2 < 2; ++mt2) {
      const _Float16* prow = ps + (h * NTOK + m0 + mt2 * 16 + l15) * NTOK;
      v16h a0 = load_a_frag(prow, 0, lane);
      v16h a1 = load_a_frag(prow, 32, lane);
#pragma unroll
      for (int dt = 0; dt < 2; ++dt) {
        const int dcol = dt * 16 + l15;
        v8f acc = {};
        acc = wmma_f16(a0, *(const v16h*)(vts + (h * HDIM + dcol) * NTOK + 0 + koff2), acc);
        acc = wmma_f16(a1, *(const v16h*)(vts + (h * HDIM + dcol) * NTOK + 32 + koff2), acc);
#pragma unroll
        for (int r = 0; r < 8; ++r) {
          const int tok = m0 + mt2 * 16 + r + lhi8;
          aos[tok * CCH + h * HDIM + dcol] = (_Float16)acc[r];
        }
      }
    }
  }
  __syncthreads();   // ps region dead from here -> reuse as f32 out staging

  // ---- phase 5: out = aos @ Wproj + b -> f32 staging in LDS
  {
    const int mt = wave & 3;
    const int m0 = mt * 16;
    const _Float16* arow = aos + (m0 + l15) * CCH;
    const v16h a0 = load_a_frag(arow, 0,  lane);
    const v16h a1 = load_a_frag(arow, 32, lane);
    const v16h a2 = load_a_frag(arow, 64, lane);
    const v16h a3 = load_a_frag(arow, 96, lane);
    for (int i = 0; i < 4; ++i) {
      const int nt = (wave >> 2) + 2 * i;   // 0..7 across waves
      const _Float16* bp = projB + (((size_t)nt * 4) * 32 + lane) * 16;
      v8f acc = {};
      acc = wmma_f16(a0, *(const v16h*)(bp + 0 * 512), acc);
      acc = wmma_f16(a1, *(const v16h*)(bp + 1 * 512), acc);
      acc = wmma_f16(a2, *(const v16h*)(bp + 2 * 512), acc);
      acc = wmma_f16(a3, *(const v16h*)(bp + 3 * 512), acc);
      if (i + 1 < 4)
        __builtin_prefetch(bp + 4 * 512, 0, 3);
      const int   n  = nt * 16 + l15;
      const float pb = proj_b[n];
#pragma unroll
      for (int r = 0; r < 8; ++r)
        outs[(m0 + r + lhi8) * CCH + n] = acc[r] + pb;
    }
  }
  __syncthreads();

  // ---- phase 6: coalesced f32 writeout with roll(+S): dst=(g+S)&511
  for (int i = tid; i < NTOK * (CCH / 4); i += 256) {
    int t  = i >> 5;
    int c4 = i & 31;
    int gy = (wy * WINSZ + (t >> 3) + SHIFTV) & (IMG - 1);
    int gx = (wx * WINSZ + (t & 7) + SHIFTV) & (IMG - 1);
    *(float4*)(out + (((size_t)bb * IMG + gy) * IMG + gx) * CCH + c4 * 4) =
        *(const float4*)(outs + t * CCH + c4 * 4);
  }
}

extern "C" void kernel_launch(void* const* d_in, const int* in_sizes, int n_in,
                              void* d_out, int out_size, void* d_ws, size_t ws_size,
                              hipStream_t stream) {
  (void)in_sizes; (void)n_in; (void)out_size; (void)ws_size;
  const float* x        = (const float*)d_in[0];
  const float* qkv_w    = (const float*)d_in[1];
  const float* qkv_b    = (const float*)d_in[2];
  const float* proj_w   = (const float*)d_in[3];
  const float* proj_b   = (const float*)d_in[4];
  const float* bias_tab = (const float*)d_in[5];

  _Float16* qkvB = (_Float16*)d_ws;
  _Float16* projB = qkvB + QKV_HALF;

  // 1) repack tiny weights into WMMA B-fragment streams (L2-resident thereafter)
  repack_weights<<<(QKV_HALF + PRJ_HALF) / 256, 256, 0, stream>>>(qkv_w, proj_w,
                                                                  qkvB, projB);
  // 2) one workgroup per 8x8 window: 4 * 64 * 64 = 16384 windows
  swin_attn_fused<<<4 * 64 * 64, 256, 0, stream>>>(x, qkv_b, proj_b, bias_tab,
                                                   qkvB, projB, (float*)d_out);
}
